// AttentionFusion_26173530702687
// MI455X (gfx1250) — compile-verified, matrix operand order checked by reading, not run on hardware
//
#include <hip/hip_runtime.h>

typedef _Float16 v8h  __attribute__((ext_vector_type(8)));
typedef _Float16 v16h __attribute__((ext_vector_type(16)));
typedef float    v8f  __attribute__((ext_vector_type(8)));
typedef unsigned int v4u __attribute__((ext_vector_type(4)));
typedef unsigned int v8u __attribute__((ext_vector_type(8)));

#define B_   8
#define C_   128
#define H_   256
#define W_   256
#define HD_  64
#define N_   4096   // HD_*HD_

// Build a 16x16x32 WMMA A/B operand (v16h) from two contiguous 8-half runs in LDS.
// K-index layout per lane-half h: regs 0-3 hold K = {0..7}+8h, regs 4-7 hold K = {16..23}+8h.
__device__ __forceinline__ v16h ld_ab(const _Float16* lo, const _Float16* hi) {
  v8h a = *(const v8h*)lo;
  v8h b = *(const v8h*)hi;
  return __builtin_shufflevector(a, b, 0,1,2,3,4,5,6,7,8,9,10,11,12,13,14,15);
}

// max-reduce across the 16-lane DPP row (wave32: two independent rows) -- pure VALU.
template <int CTRL>
__device__ __forceinline__ float dpp_max_step(float x) {
  int y = __builtin_amdgcn_update_dpp(__float_as_int(x), __float_as_int(x),
                                      CTRL, 0xF, 0xF, true);
  return fmaxf(x, __int_as_float(y));
}
__device__ __forceinline__ float rowmax16(float x) {
  x = dpp_max_step<0x121>(x);  // row_ror:1
  x = dpp_max_step<0x122>(x);  // row_ror:2
  x = dpp_max_step<0x124>(x);  // row_ror:4
  x = dpp_max_step<0x128>(x);  // row_ror:8
  return x;
}

// ---------------------------------------------------------------------------
// Stage 1: fused 4x bilinear downsample (exact: mean of 2x2 block at 4i+1..4i+2)
// + fp32->f16, writing dst[b][n][c] (c contiguous).  grid=(HD_,B_,2), block=256
// ---------------------------------------------------------------------------
__global__ void __launch_bounds__(256)
downsample_kernel(const float* __restrict__ wide, const float* __restrict__ narrow,
                  _Float16* __restrict__ q_t, _Float16* __restrict__ k_t) {
  __shared__ _Float16 tile[64][136];
  const int y = blockIdx.x;
  const int b = blockIdx.y;
  const float* src = blockIdx.z ? narrow : wide;
  _Float16*    dst = blockIdx.z ? k_t    : q_t;
  const int tid = threadIdx.x;
  const int x  = tid & 63;
  const int c0 = tid >> 6;
  const int iy = 4 * y + 1;
  const int ix = 4 * x + 1;
  const float* base = src + ((((size_t)b * C_) * H_ + iy) * W_ + ix);
  #pragma unroll 4
  for (int cc = 0; cc < C_; cc += 4) {
    const int c = cc + c0;
    const float* p = base + (size_t)c * (H_ * W_);
    float v = 0.25f * (p[0] + p[1] + p[W_] + p[W_ + 1]);
    tile[x][c] = (_Float16)v;
  }
  __syncthreads();
  uint4* od = (uint4*)(dst + ((size_t)b * N_ + (size_t)y * 64) * C_);
  for (int i = tid; i < (64 * C_) / 8; i += 256) {
    const int xx = i >> 4;
    const int cg = (i & 15) * 8;
    od[i] = *(const uint4*)&tile[xx][cg];
  }
}

// ---------------------------------------------------------------------------
// Stage 2: flash attention. 8 waves/block, each wave owns 16 n-rows.
// K tiles are DMA'd into LDS by the Tensor Data Mover with hardware row
// padding matching the bank-conflict-free [32][136] layout.
// grid = (N_/128, B_), block = 256
// ---------------------------------------------------------------------------
__global__ void __launch_bounds__(256)
attn_kernel(const _Float16* __restrict__ q_t, const _Float16* __restrict__ k_t,
            float* __restrict__ o_t) {
  __shared__ _Float16 Qs [128][136];   // Q rows [n][c]            34816 B
  __shared__ _Float16 Kts[ 32][136];   // K tile [m][c] (TDM dest)  8704 B
  __shared__ _Float16 Ks2[128][ 40];   // K tile [c][m]            10240 B
  __shared__ _Float16 Ps [  8][16][40];// per-wave P transpose     10240 B => 64000 B

  const int b    = blockIdx.y;
  const int n0   = blockIdx.x * 128;
  const int tid  = threadIdx.x;
  const int lane = tid & 31;
  const int wv   = tid >> 5;
  const int col  = lane & 15;
  const int h    = lane >> 4;
  const int rowL = wv * 16 + col;

  // Load the block's 128x128 Q slab to LDS (coalesced b128).
  const uint4* qsrc = (const uint4*)(q_t + ((size_t)b * N_ + n0) * C_);
  for (int i = tid; i < 2048; i += 256) {
    const int r = i >> 4, c = (i & 15) * 8;
    *(uint4*)&Qs[r][c] = qsrc[i];
  }
  __syncthreads();

  v16h qa[4];
  #pragma unroll
  for (int kc = 0; kc < 4; ++kc)
    qa[kc] = ld_ab(&Qs[rowL][kc * 32 + h * 8], &Qs[rowL][kc * 32 + 16 + h * 8]);

  v16h onesb;                       // all-ones B: row-sum accumulator feed
  #pragma unroll
  for (int j = 0; j < 16; ++j) onesb[j] = (_Float16)1.0f;

  v8f o[8], osum;                   // O tiles + row-sum tile (flash normalizer)
  #pragma unroll
  for (int ct = 0; ct < 8; ++ct)
    #pragma unroll
    for (int v = 0; v < 8; ++v) o[ct][v] = 0.0f;
  #pragma unroll
  for (int v = 0; v < 8; ++v) osum[v] = 0.0f;
  float rmax[8];
  #pragma unroll
  for (int v = 0; v < 8; ++v) rmax[v] = -__builtin_inff();

  // Loop-invariant pieces of the TDM descriptor (ISA 08_async_tensor §8.3-8.4).
  const unsigned kts_lds = (unsigned)(uintptr_t)(&Kts[0][0]); // flat low 32 = LDS offset
  // group1: data_size=2B(code1), pad_enable, pad_interval=64DW(code5), pad_amount=4DW(code3)
  //         tensor_dim0=128, tensor_dim1=1<<20, tile=128x32, dim0_stride=128
  const v8u g1 = { (1u << 16) | (1u << 20) | (5u << 22) | (3u << 25),
                   128u << 16, 0u, 0x10u | (128u << 16), 32u, 128u, 0u, 0u };
  const v4u gz = { 0u, 0u, 0u, 0u };
  const int wv_u = __builtin_amdgcn_readfirstlane(wv);  // scalar branch: TDM ignores EXEC

  const float scale = 0.08838834764831845f;  // 1/sqrt(128)

  for (int mt = 0; mt < N_ / 32; ++mt) {
    __syncthreads();   // all waves done reading previous Kts/Ks2
    const _Float16* ktile = k_t + ((size_t)b * N_ + (size_t)mt * 32) * C_;
    if (wv_u == 0) {
      // Tensor Data Mover: 32x128 f16 tile -> LDS with 8-half row padding.
      const unsigned long long ga = (unsigned long long)(uintptr_t)ktile;
      v4u g0 = { 1u /*count=1*/, kts_lds,
                 (unsigned)ga, (unsigned)(ga >> 32) | 0x80000000u /*type=2*/ };
      asm volatile("tensor_load_to_lds %0, %1, %2, %3"
                   :
                   : "s"(g0), "s"(g1), "s"(gz), "s"(gz)
                   : "memory");
      __builtin_amdgcn_s_wait_tensorcnt(0);
    } else if (wv_u == 1 && mt + 1 < N_ / 32) {
      // warm L2 for the next tile while wave0 drives the TDM
      __builtin_prefetch((const char*)(ktile) + 8192 + lane * 256, 0, 1);
    }
    __syncthreads();   // Kts ready for everyone

    // Build [c][m] copy for GEMM2's B operand (LDS->LDS, conflict-padded).
    for (int i = tid; i < 512; i += 256) {
      const int m = i >> 4, cg = (i & 15) * 8;
      union { uint4 u; _Float16 f[8]; } cv;
      cv.u = *(const uint4*)&Kts[m][cg];
      #pragma unroll
      for (int j = 0; j < 8; ++j) Ks2[cg + j][m] = cv.f[j];
    }
    __syncthreads();   // Ks2 ready

    // ---- GEMM1: S[16x32] = Q^T(16x128) * K(128x32) ----
    // Preload all 8 B operands into distinct registers so the DS loads
    // pipeline instead of forcing s_wait_dscnt 0 before every WMMA.
    v16h bq[8];
    #pragma unroll
    for (int kc = 0; kc < 4; ++kc) {
      bq[kc]     = ld_ab(&Kts[col     ][kc * 32 + h * 8], &Kts[col     ][kc * 32 + 16 + h * 8]);
      bq[4 + kc] = ld_ab(&Kts[16 + col][kc * 32 + h * 8], &Kts[16 + col][kc * 32 + 16 + h * 8]);
    }
    v8f s0, s1;
    #pragma unroll
    for (int v = 0; v < 8; ++v) { s0[v] = 0.0f; s1[v] = 0.0f; }
    #pragma unroll
    for (int kc = 0; kc < 4; ++kc) {
      s0 = __builtin_amdgcn_wmma_f32_16x16x32_f16(false, qa[kc], false, bq[kc],     (short)0, s0, false, false);
      s1 = __builtin_amdgcn_wmma_f32_16x16x32_f16(false, qa[kc], false, bq[4 + kc], (short)0, s1, false, false);
    }

    // Issue GEMM2's 8 B operand loads now: independent of softmax, their
    // latency hides under the VALU-heavy softmax below.
    v16h kb[8];
    #pragma unroll
    for (int ct = 0; ct < 8; ++ct)
      kb[ct] = ld_ab(&Ks2[ct * 16 + col][h * 8], &Ks2[ct * 16 + col][16 + h * 8]);

    // ---- online softmax: DPP row-max, exp, P -> LDS (A-layout transpose) ----
    float alpha[8];
    #pragma unroll
    for (int v = 0; v < 8; ++v) {
      const float a0 = s0[v] * scale, a1 = s1[v] * scale;
      const float mx = rowmax16(fmaxf(a0, a1));
      const float nm = fmaxf(rmax[v], mx);
      alpha[v] = __expf(rmax[v] - nm);
      rmax[v] = nm;
      const float p0 = __expf(a0 - nm), p1 = __expf(a1 - nm);
      Ps[wv][v + 8 * h][col]      = (_Float16)p0;
      Ps[wv][v + 8 * h][16 + col] = (_Float16)p1;
    }
    #pragma unroll
    for (int ct = 0; ct < 8; ++ct)
      #pragma unroll
      for (int v = 0; v < 8; ++v) o[ct][v] *= alpha[v];
    #pragma unroll
    for (int v = 0; v < 8; ++v) osum[v] *= alpha[v];

    // ---- GEMM2: O[16x128] += P*K^T; row sums ride along as a 9th WMMA ----
    v16h pa = ld_ab(&Ps[wv][col][h * 8], &Ps[wv][col][16 + h * 8]);
    osum = __builtin_amdgcn_wmma_f32_16x16x32_f16(false, pa, false, onesb, (short)0, osum, false, false);
    #pragma unroll
    for (int ct = 0; ct < 8; ++ct)
      o[ct] = __builtin_amdgcn_wmma_f32_16x16x32_f16(false, pa, false, kb[ct], (short)0, o[ct], false, false);
  }

  // normalize by the WMMA-accumulated row sums and store O to o_t[b][n][c]
  float rinv[8];
  #pragma unroll
  for (int v = 0; v < 8; ++v) rinv[v] = 1.0f / osum[v];
  const size_t nb = (size_t)b * N_ + n0 + wv * 16;
  #pragma unroll
  for (int ct = 0; ct < 8; ++ct)
    #pragma unroll
    for (int v = 0; v < 8; ++v)
      o_t[(nb + v + 8 * h) * C_ + ct * 16 + col] = o[ct][v] * rinv[v];
}

// ---------------------------------------------------------------------------
// Stage 3: 4x bilinear upsample (align_corners=False) + residual add.
// grid = (H_, B_), block = 256
// ---------------------------------------------------------------------------
__global__ void __launch_bounds__(256)
up_add_kernel(const float* __restrict__ wide, const float* __restrict__ o_t,
              float* __restrict__ out) {
  __shared__ _Float16 so[2][64][130];
  const int y = blockIdx.x, b = blockIdx.y, tid = threadIdx.x;

  const float sy = 0.25f * (float)y - 0.375f;
  const float fy = sy - floorf(sy);
  const int   y0 = (int)floorf(sy);
  const int  iy0 = min(max(y0, 0), 63);
  const int  iy1 = min(max(y0 + 1, 0), 63);

  const float* r0 = o_t + ((size_t)b * N_ + (size_t)iy0 * 64) * C_;
  const float* r1 = o_t + ((size_t)b * N_ + (size_t)iy1 * 64) * C_;
  for (int i = tid; i < 64 * C_; i += 256) {
    const int n = i >> 7, c = i & 127;
    so[0][n][c] = (_Float16)r0[i];
    so[1][n][c] = (_Float16)r1[i];
  }
  __syncthreads();

  const int   x  = tid;
  const float sx = 0.25f * (float)x - 0.375f;
  const float fx = sx - floorf(sx);
  const int   x0 = (int)floorf(sx);
  const int  ix0 = min(max(x0, 0), 63);
  const int  ix1 = min(max(x0 + 1, 0), 63);
  const float w00 = (1.0f - fy) * (1.0f - fx), w01 = (1.0f - fy) * fx;
  const float w10 = fy * (1.0f - fx),          w11 = fy * fx;

  const size_t base = (((size_t)b * C_) * H_ + y) * W_ + x;
  #pragma unroll 4
  for (int c = 0; c < C_; ++c) {
    const float v = w00 * (float)so[0][ix0][c] + w01 * (float)so[0][ix1][c]
                  + w10 * (float)so[1][ix0][c] + w11 * (float)so[1][ix1][c];
    const size_t idx = base + (size_t)c * (H_ * W_);
    out[idx] = wide[idx] + v;
  }
}

// ---------------------------------------------------------------------------
extern "C" void kernel_launch(void* const* d_in, const int* in_sizes, int n_in,
                              void* d_out, int out_size, void* d_ws, size_t ws_size,
                              hipStream_t stream) {
  const float* wide   = (const float*)d_in[0];
  const float* narrow = (const float*)d_in[1];
  float*       out    = (float*)d_out;

  // workspace: q_t (8 MiB f16) | k_t (8 MiB f16) | o_t (16 MiB f32) = 32 MiB
  char* ws = (char*)d_ws;
  const size_t half_bytes = (size_t)B_ * N_ * C_ * sizeof(_Float16);
  _Float16* q_t = (_Float16*)ws;
  _Float16* k_t = (_Float16*)(ws + half_bytes);
  float*    o_t = (float*)(ws + 2 * half_bytes);

  downsample_kernel<<<dim3(HD_, B_, 2), 256, 0, stream>>>(wide, narrow, q_t, k_t);
  attn_kernel      <<<dim3(N_ / 128, B_), 256, 0, stream>>>(q_t, k_t, o_t);
  up_add_kernel    <<<dim3(H_, B_), 256, 0, stream>>>(wide, o_t, out);
}
